// GCNNet_23295902613895
// MI455X (gfx1250) — compile-verified, matrix operand order checked by reading, not run on hardware
//
#include <hip/hip_runtime.h>
#include <hip/hip_bf16.h>
#include <math.h>

typedef __bf16 bf16_t;
typedef __attribute__((ext_vector_type(16))) __bf16 v16bf;
typedef __attribute__((ext_vector_type(8)))  __bf16 v8bf;
typedef __attribute__((ext_vector_type(8)))  float  v8f;

__device__ __forceinline__ bf16_t f2bf(float x) {
    unsigned u = __float_as_uint(x);
    unsigned r = u + 0x7FFFu + ((u >> 16) & 1u);   // round to nearest even
    union { unsigned short s; bf16_t b; } cv;
    cv.s = (unsigned short)(r >> 16);
    return cv.b;
}

// ---------------- utility ----------------
__global__ void k_fill(float* __restrict__ p, float v, int n) {
    int i = blockIdx.x * blockDim.x + threadIdx.x;
    if (i < n) p[i] = v;
}

__global__ void k_copy(const float* __restrict__ s, float* __restrict__ d, int n) {
    int i = blockIdx.x * blockDim.x + threadIdx.x;
    if (i < n) d[i] = s[i];
}

// ---------------- degree / normalization ----------------
__global__ void k_edge_deg(const int* __restrict__ dst, float* __restrict__ deg, int E) {
    int e = blockIdx.x * blockDim.x + threadIdx.x;
    if (e < E) atomicAdd(&deg[dst[e]], 1.0f);
}

__global__ void k_node_norm(const float* __restrict__ deg, float* __restrict__ dis,
                            float* __restrict__ scoef, int N) {
    int i = blockIdx.x * blockDim.x + threadIdx.x;
    if (i < N) {
        float d = deg[i];
        dis[i]   = rsqrtf(d);
        scoef[i] = 1.0f / d;
    }
}

__global__ void k_ecoef(const int* __restrict__ src, const int* __restrict__ dst,
                        const float* __restrict__ dis, float* __restrict__ ecoef, int E) {
    int e = blockIdx.x * blockDim.x + threadIdx.x;
    if (e < E) ecoef[e] = dis[src[e]] * dis[dst[e]];
}

// ---------------- fp32 -> bf16 (zero-padded to Kp, Mpad) ----------------
__global__ void k_cvt_bf16(const float* __restrict__ X, bf16_t* __restrict__ A,
                           int M, int Mpad, int fin, int Kp) {
    int i = blockIdx.x * blockDim.x + threadIdx.x;
    int total = Mpad * Kp;
    if (i >= total) return;
    int row = i / Kp, k = i - row * Kp;
    float v = (row < M && k < fin) ? X[(size_t)row * fin + k] : 0.0f;
    A[i] = f2bf(v);
}

// ---------------- pack W[K x fout] into per-lane WMMA B fragments ----------------
// Fragment (nt,ks): 32 lanes x 16 bf16, lane-contiguous. Lane l holds column
// n0+(l&15); element e maps to K = k0 + (e<8?0:16) + ((l>=16)?8:0) + (e&7).
__global__ void k_pack_B(const float* __restrict__ W, bf16_t* __restrict__ Bp,
                         int fin, int fout, int Kp) {
    int ksteps = Kp >> 5;
    int total  = (fout >> 4) * ksteps * 512;
    int i = blockIdx.x * blockDim.x + threadIdx.x;
    if (i >= total) return;
    int e    = i & 15;
    int lane = (i >> 4) & 31;
    int frag = i >> 9;
    int ks = frag % ksteps;
    int nt = frag / ksteps;
    int kl = ((e < 8) ? 0 : 16) + ((lane >> 4) << 3) + (e & 7);
    int k  = (ks << 5) + kl;
    int n  = (nt << 4) + (lane & 15);
    float v = (k < fin) ? W[(size_t)k * fout + n] : 0.0f;
    Bp[i] = f2bf(v);
}

// ---------------- WMMA GEMM: C[M x fout] = A[M x Kp] * W ----------------
// One wave per 16-row M tile, loops over 16-col N tiles, K in steps of 32.
__global__ void k_wmma_gemm(const bf16_t* __restrict__ A, const bf16_t* __restrict__ Bp,
                            float* __restrict__ C, int mtiles, int Kp, int fout) {
    const int lane = threadIdx.x & 31;
    const int wave = blockIdx.x * (blockDim.x >> 5) + (threadIdx.x >> 5);
    if (wave >= mtiles) return;                      // wave-uniform: EXEC all-ones inside
    const int ksteps = Kp >> 5;
    const int ntiles = fout >> 4;
    const int m0   = wave << 4;
    const int koff = (lane >> 4) << 3;               // 0 or 8 (K half per lane group)
    const bf16_t* arow = A + (size_t)(m0 + (lane & 15)) * Kp;
    const int mstore = m0 + ((lane >> 4) << 3);      // C/D layout: lanes 16-31 -> M+8
    const int cstore = lane & 15;

    for (int nt = 0; nt < ntiles; ++nt) {
        v8f acc = {0.f, 0.f, 0.f, 0.f, 0.f, 0.f, 0.f, 0.f};
        const bf16_t* bptr = Bp + (size_t)nt * ksteps * 512 + lane * 16;
        for (int ks = 0; ks < ksteps; ++ks) {
            const int k0 = ks << 5;
            v8bf lo = *reinterpret_cast<const v8bf*>(arow + k0 + koff);        // K[koff..+7]
            v8bf hi = *reinterpret_cast<const v8bf*>(arow + k0 + 16 + koff);   // K[16+koff..]
            v16bf a;
#pragma unroll
            for (int j = 0; j < 8; ++j) { a[j] = lo[j]; a[j + 8] = hi[j]; }
            v16bf b = *reinterpret_cast<const v16bf*>(bptr);
            bptr += 512;
            acc = __builtin_amdgcn_wmma_f32_16x16x32_bf16(false, a, false, b,
                                                          (short)0, acc, false, false);
        }
        float* cb = C + (size_t)mstore * fout + nt * 16 + cstore;
#pragma unroll
        for (int v = 0; v < 8; ++v) cb[(size_t)v * fout] = acc[v];
    }
}

// ---------------- edge scatter-add: AGG[dst] += HW[src] * ecoef ----------------
__global__ void k_edge_agg(const float* __restrict__ HW, const int* __restrict__ src,
                           const int* __restrict__ dst, const float* __restrict__ ecoef,
                           float* __restrict__ AGG, int E, int fout, int lc) {
    int i = blockIdx.x * blockDim.x + threadIdx.x;
    int total = E << lc;                    // lc = log2(fout/4)
    if (i >= total) return;
    int e  = i >> lc;
    int c4 = (i & ((1 << lc) - 1)) << 2;
    float w = ecoef[e];
    float4 v = *reinterpret_cast<const float4*>(HW + (size_t)src[e] * fout + c4);
    float* o = AGG + (size_t)dst[e] * fout + c4;
    atomicAdd(o + 0, v.x * w);
    atomicAdd(o + 1, v.y * w);
    atomicAdd(o + 2, v.z * w);
    atomicAdd(o + 3, v.w * w);
}

// ---------------- combine (+self-loop, +bias) and per-column partial stats ----
// blockDim.x == fout; block handles 256 rows; coalesced row-major reads.
// T may alias AGG_in (element read once then written by same thread).
__global__ void k_combine_stats(const float* __restrict__ HW, const float* AGG_in,
                                const float* __restrict__ scoef, const float* __restrict__ bias,
                                float* T, float* __restrict__ sum, float* __restrict__ sumsq,
                                int M, int fout) {
    int col = threadIdx.x;
    int r0 = blockIdx.x * 256;
    int r1 = r0 + 256; if (r1 > M) r1 = M;
    float b = bias[col];
    float s = 0.f, ss = 0.f;
    for (int r = r0; r < r1; ++r) {
        size_t i = (size_t)r * fout + col;
        float t = HW[i];
        if (scoef)  t *= scoef[r];
        if (AGG_in) t += AGG_in[i];
        t += b;
        T[i] = t;
        s += t; ss += t * t;
    }
    atomicAdd(&sum[col], s);
    atomicAdd(&sumsq[col], ss);
}

__global__ void k_bn_finalize(const float* __restrict__ sum, const float* __restrict__ sumsq,
                              const float* __restrict__ g, const float* __restrict__ beta,
                              float* __restrict__ scale, float* __restrict__ shift,
                              int M, int fout) {
    int c = blockIdx.x * blockDim.x + threadIdx.x;
    if (c >= fout) return;
    float inv_m = 1.0f / (float)M;
    float m = sum[c] * inv_m;
    float v = sumsq[c] * inv_m - m * m;
    float inv = rsqrtf(fmaxf(v, 0.0f) + 1e-5f);
    float a = g[c] * inv;
    scale[c] = a;
    shift[c] = beta[c] - a * m;
}

__global__ void k_bn_relu(const float* __restrict__ T, const float* __restrict__ scale,
                          const float* __restrict__ shift, float* __restrict__ H,
                          int total, int fout) {
    int i = blockIdx.x * blockDim.x + threadIdx.x;
    if (i >= total) return;
    int col = i & (fout - 1);               // fout is a power of two
    H[i] = fmaxf(fmaf(T[i], scale[col], shift[col]), 0.0f);
}

// ---------------- global max pool (values >= 0 after ReLU) ----------------
__global__ void k_pool_max(const float* __restrict__ H, const int* __restrict__ batch,
                           unsigned* __restrict__ pooled, int N) {
    int i = blockIdx.x * blockDim.x + threadIdx.x;
    if (i >= N * 32) return;
    int row = i >> 5, col = i & 31;
    atomicMax(&pooled[(size_t)batch[row] * 32 + col], __float_as_uint(H[i]));
}

// ---------------- final 256->1 + sigmoid ----------------
__global__ void k_head(const float* __restrict__ G, const float* __restrict__ W,
                       const float* __restrict__ b, float* __restrict__ out,
                       int rows, int K) {
    int r = blockIdx.x * blockDim.x + threadIdx.x;
    if (r >= rows) return;
    float acc = b[0];
    const float* g = G + (size_t)r * K;
    for (int k = 0; k < K; ++k) acc = fmaf(g[k], W[k], acc);
    out[r] = 1.0f / (1.0f + __expf(-acc));
}

extern "C" void kernel_launch(void* const* d_in, const int* in_sizes, int n_in,
                              void* d_out, int out_size, void* d_ws, size_t ws_size,
                              hipStream_t stream) {
    const float* x    = (const float*)d_in[0];
    const int* eidx   = (const int*)d_in[1];
    const int* batch  = (const int*)d_in[2];
    const int N = in_sizes[0] / 78;
    const int E = in_sizes[1] / 2;
    const int* src = eidx;
    const int* dst = eidx + E;
    const int NG = 512;

    const float *cW[4], *cbv[4], *cg[4], *cbeta[4];
    for (int l = 0; l < 4; ++l) {
        cW[l]    = (const float*)d_in[3 + 4 * l];
        cbv[l]   = (const float*)d_in[4 + 4 * l];
        cg[l]    = (const float*)d_in[5 + 4 * l];
        cbeta[l] = (const float*)d_in[6 + 4 * l];
    }
    const float *fW[4], *fbv[4], *fg[3], *fbeta[3];
    for (int l = 0; l < 3; ++l) {
        fW[l]    = (const float*)d_in[19 + 4 * l];
        fbv[l]   = (const float*)d_in[20 + 4 * l];
        fg[l]    = (const float*)d_in[21 + 4 * l];
        fbeta[l] = (const float*)d_in[22 + 4 * l];
    }
    fW[3]  = (const float*)d_in[31];
    fbv[3] = (const float*)d_in[32];

    const int Mpad = ((N + 15) / 16) * 16;
    char* base = (char*)d_ws;
    size_t off = 0;
    auto carve = [&](size_t bytes) -> void* {
        void* p = base + off;
        off = (off + bytes + 255) & ~(size_t)255;
        return p;
    };
    float*  deg    = (float*)carve((size_t)N * 4);
    float*  dis    = (float*)carve((size_t)N * 4);
    float*  scoef  = (float*)carve((size_t)N * 4);
    float*  ecoef  = (float*)carve((size_t)E * 4);
    bf16_t* Abf    = (bf16_t*)carve((size_t)Mpad * 256 * 2);
    bf16_t* Bp     = (bf16_t*)carve((size_t)2 << 20);
    float*  HW     = (float*)carve((size_t)Mpad * 256 * 4);   // GEMM out, also post-BN H
    float*  AGG    = (float*)carve((size_t)Mpad * 256 * 4);   // scatter accum / pre-BN T
    float*  sum    = (float*)carve(1024 * 4);
    float*  sumsq  = (float*)carve(1024 * 4);
    float*  scale  = (float*)carve(1024 * 4);
    float*  shift  = (float*)carve(1024 * 4);
    float*  pooled = (float*)carve((size_t)NG * 32 * 4);
    (void)ws_size; (void)n_in; (void)out_size;

    auto cdiv = [](int a, int b) { return (a + b - 1) / b; };

    // degree = in-degree + 1 (self loop); dis = deg^-1/2; scoef = 1/deg; ecoef per edge
    k_fill<<<cdiv(N, 256), 256, 0, stream>>>(deg, 1.0f, N);
    k_edge_deg<<<cdiv(E, 256), 256, 0, stream>>>(dst, deg, E);
    k_node_norm<<<cdiv(N, 256), 256, 0, stream>>>(deg, dis, scoef, N);
    k_ecoef<<<cdiv(E, 256), 256, 0, stream>>>(src, dst, dis, ecoef, E);

    // ---- 4 GCN conv layers ----
    const int cfin[4]  = {78, 256, 128, 64};
    const int cfout[4] = {256, 128, 64, 32};
    const float* hin = x;
    for (int l = 0; l < 4; ++l) {
        const int fin = cfin[l], fout = cfout[l];
        const int Kp = ((fin + 31) / 32) * 32;
        const int mtiles = Mpad / 16;
        k_cvt_bf16<<<cdiv(Mpad * Kp, 256), 256, 0, stream>>>(hin, Abf, N, Mpad, fin, Kp);
        int btotal = (fout / 16) * (Kp / 32) * 512;
        k_pack_B<<<cdiv(btotal, 256), 256, 0, stream>>>(cW[l], Bp, fin, fout, Kp);
        k_wmma_gemm<<<cdiv(mtiles, 8), 256, 0, stream>>>(Abf, Bp, HW, mtiles, Kp, fout);

        k_fill<<<cdiv(N * fout, 256), 256, 0, stream>>>(AGG, 0.0f, N * fout);
        int lc = 0; { int c = fout >> 2; while ((1 << lc) < c) ++lc; }
        k_edge_agg<<<cdiv(E << lc, 256), 256, 0, stream>>>(HW, src, dst, ecoef, AGG, E, fout, lc);

        k_fill<<<4, 256, 0, stream>>>(sum, 0.0f, 1024);
        k_fill<<<4, 256, 0, stream>>>(sumsq, 0.0f, 1024);
        k_combine_stats<<<cdiv(N, 256), fout, 0, stream>>>(HW, AGG, scoef, cbv[l],
                                                           AGG, sum, sumsq, N, fout);
        k_bn_finalize<<<cdiv(fout, 256), 256, 0, stream>>>(sum, sumsq, cg[l], cbeta[l],
                                                           scale, shift, N, fout);
        k_bn_relu<<<cdiv(N * fout, 256), 256, 0, stream>>>(AGG, scale, shift, HW,
                                                           N * fout, fout);
        hin = HW;
    }

    // ---- global max pool over graphs ----
    k_fill<<<cdiv(NG * 32, 256), 256, 0, stream>>>(pooled, 0.0f, NG * 32);
    k_pool_max<<<cdiv(N * 32, 256), 256, 0, stream>>>(HW, batch, (unsigned*)pooled, N);

    // ---- FC layers (reuse GEMM/BN path, M = 512) ----
    const int ffin[3]  = {32, 1024, 512};
    const int ffout[3] = {1024, 512, 256};
    const float* gin = pooled;
    for (int l = 0; l < 3; ++l) {
        const int fin = ffin[l], fout = ffout[l], Kp = fin;
        const int mtiles = NG / 16;
        k_cvt_bf16<<<cdiv(NG * Kp, 256), 256, 0, stream>>>(gin, Abf, NG, NG, fin, Kp);
        int btotal = (fout / 16) * (Kp / 32) * 512;
        k_pack_B<<<cdiv(btotal, 256), 256, 0, stream>>>(fW[l], Bp, fin, fout, Kp);
        k_wmma_gemm<<<cdiv(mtiles, 8), 256, 0, stream>>>(Abf, Bp, HW, mtiles, Kp, fout);

        k_fill<<<4, 256, 0, stream>>>(sum, 0.0f, 1024);
        k_fill<<<4, 256, 0, stream>>>(sumsq, 0.0f, 1024);
        k_combine_stats<<<cdiv(NG, 256), fout, 0, stream>>>(HW, nullptr, nullptr, fbv[l],
                                                            AGG, sum, sumsq, NG, fout);
        k_bn_finalize<<<cdiv(fout, 256), 256, 0, stream>>>(sum, sumsq, fg[l], fbeta[l],
                                                           scale, shift, NG, fout);
        k_bn_relu<<<cdiv(NG * fout, 256), 256, 0, stream>>>(AGG, scale, shift, HW,
                                                            NG * fout, fout);
        gin = HW;
    }

    // ---- head: sigmoid(g @ fW3 + fb3) and last_layer_features ----
    float* out = (float*)d_out;
    k_head<<<2, 256, 0, stream>>>(HW, fW[3], fbv[3], out, NG, 256);
    k_copy<<<cdiv(NG * 256, 256), 256, 0, stream>>>(HW, out + NG, NG * 256);
}
